// S2V_QN_2_84808424227518
// MI455X (gfx1250) — matrix operand order repaired and checked by reading, not compiled, not run on hardware
//
#include <hip/hip_runtime.h>
#include <hip/hip_bf16.h>

#define N_NODES 8192
#define D_DIM   64
#define B_BATCH 2

typedef __attribute__((ext_vector_type(16))) __bf16  v16bf;
typedef __attribute__((ext_vector_type(2)))  __bf16  v2bf;
typedef __attribute__((ext_vector_type(8)))  float    v8f;
typedef __attribute__((ext_vector_type(4)))  unsigned u4v;

union BV { v16bf v; unsigned u[8]; u4v q[2]; };

// Hardware f32->bf16 (lowers to v_cvt_pk_bf16_f32 on gfx1250, RNE).
__device__ __forceinline__ unsigned short f2bf(float f) {
  return __builtin_bit_cast(unsigned short, (__bf16)f);
}
__device__ __forceinline__ unsigned pk2(float a, float b) {
  v2bf p;
  p[0] = (__bf16)a;
  p[1] = (__bf16)b;
  return __builtin_bit_cast(unsigned, p);
}
// Pack two contiguous 8-float runs into the 16-bit A-matrix lane layout
// (VGPR0..3 = run0 pairs, VGPR4..7 = run1 pairs).
__device__ __forceinline__ void packA(BV& A, const float4& a0, const float4& a1,
                                      const float4& b0, const float4& b1) {
  A.u[0] = pk2(a0.x, a0.y); A.u[1] = pk2(a0.z, a0.w);
  A.u[2] = pk2(a1.x, a1.y); A.u[3] = pk2(a1.z, a1.w);
  A.u[4] = pk2(b0.x, b0.y); A.u[5] = pk2(b0.z, b0.w);
  A.u[6] = pk2(b1.x, b1.y); A.u[7] = pk2(b1.z, b1.w);
}

// ---------------------------------------------------------------------------
// Stage A: mu_first = relu(xv * mu_1); 2x pre Linear+ReLU; store mu_first(f32)
// and mu_pre transposed [b][d][n] as bf16 for the aggregation GEMM.
// ---------------------------------------------------------------------------
__global__ __launch_bounds__(256) void s2v_embed(
    const float* __restrict__ xv, const float* __restrict__ mu1,
    const float* __restrict__ preW, const float* __restrict__ preb,
    float* __restrict__ muFirst, unsigned short* __restrict__ muPreT)
{
  __shared__ float sW[2 * 64 * 64];
  __shared__ float sMu1[64];
  __shared__ float sB[128];
  const int tid = threadIdx.x;
  for (int i = tid; i < 8192; i += 256) sW[i] = preW[i];
  if (tid < 64)  sMu1[tid] = mu1[tid];
  if (tid < 128) sB[tid]   = preb[tid];
  __syncthreads();

  const int n = blockIdx.x * 256 + tid;        // flat node id over B*N
  const float x = xv[n];
  float m[64], t[64];
  #pragma unroll
  for (int d = 0; d < 64; ++d) { float v = x * sMu1[d]; m[d] = v > 0.f ? v : 0.f; }
  #pragma unroll
  for (int d = 0; d < 64; d += 4) {
    float4 o; o.x = m[d]; o.y = m[d+1]; o.z = m[d+2]; o.w = m[d+3];
    *(float4*)(muFirst + (size_t)n * 64 + d) = o;
  }
  for (int l = 0; l < 2; ++l) {
    const float* W  = sW + l * 4096;
    const float* bb = sB + l * 64;
    for (int j = 0; j < 64; ++j) {
      float a = bb[j];
      #pragma unroll
      for (int e = 0; e < 64; ++e) a += m[e] * W[j * 64 + e];
      t[j] = a > 0.f ? a : 0.f;
    }
    #pragma unroll
    for (int j = 0; j < 64; ++j) m[j] = t[j];
  }
  const int b2 = n >> 13, nn = n & (N_NODES - 1);
  unsigned short* dst = muPreT + (size_t)b2 * 64 * N_NODES + nn;
  #pragma unroll
  for (int d = 0; d < 64; ++d) dst[(size_t)d * N_NODES] = f2bf(m[d]);
}

// ---------------------------------------------------------------------------
// Stage B: mu_pool = adj @ mu (streaming, bf16 WMMA), fused mu_2 + 2 post
// layers (bias+ReLU) via WMMA, mu = relu(mu_first + mu2), deterministic
// per-block xv-weighted column sums for the readout.
// ---------------------------------------------------------------------------
__global__ __launch_bounds__(256) void s2v_aggregate(
    const float* __restrict__ adj, const float* __restrict__ mu2w,
    const float* __restrict__ postW, const float* __restrict__ postb,
    const float* __restrict__ xv, const float* __restrict__ muFirst,
    const unsigned short* __restrict__ muPreT,
    float* __restrict__ muOut, float* __restrict__ sPart)
{
  __shared__ __align__(16) unsigned short muT[64][72];     // mu chunk, transposed, padded
  __shared__ __align__(16) float          pool[8][16][68]; // per-wave 16x64 f32 tile
  __shared__ __align__(16) unsigned short WT[3][64][72];   // BT of mu_2 / post_W0 / post_W1
  __shared__ float sBias[2][64];
  __shared__ float wsum[2][8][64];

  const int tid  = threadIdx.x;
  const int wave = tid >> 5;
  const int lane = tid & 31;
  const int nlo  = lane & 15;
  const int h    = lane >> 4;
  const int b    = blockIdx.y;
  const int r0   = blockIdx.x * 128;

  for (int i = tid; i < 4096; i += 256) {       // weight tiles -> transposed bf16
    int j = i >> 6, e = i & 63;
    WT[0][j][e] = f2bf(mu2w[e * 64 + j]);               // B[e][j]=mu_2[e][j]
    WT[1][j][e] = f2bf(postW[j * 64 + e]);              // B[e][j]=post_W0[j][e]
    WT[2][j][e] = f2bf(postW[4096 + j * 64 + e]);       // B[e][j]=post_W1[j][e]
  }
  for (int i = tid; i < 128; i += 256) ((float*)sBias)[i] = postb[i];

  v8f acc[4] = {};
  const float* adjRow = adj + ((size_t)b * N_NODES + r0 + wave * 16 + nlo) * N_NODES;
  const unsigned short* mpt = muPreT + (size_t)b * 64 * N_NODES;
  __syncthreads();

  for (int k0 = 0; k0 < N_NODES; k0 += 64) {
    { // cooperative stage of muT[d][0..63] bf16 chunk
      int d = tid >> 2, qq = tid & 3;
      const u4v* src = (const u4v*)(mpt + (size_t)d * N_NODES + k0 + qq * 16);
      u4v s0 = src[0], s1 = src[1];
      *(u4v*)&muT[d][qq * 16]     = s0;
      *(u4v*)&muT[d][qq * 16 + 8] = s1;
    }
    if (k0 + 64 < N_NODES) __builtin_prefetch(adjRow + k0 + 64 + h * 8, 0, 3);
    __syncthreads();
    #pragma unroll
    for (int kk = 0; kk < 2; ++kk) {
      const float* p0 = adjRow + k0 + kk * 32 + h * 8;
      float4 a0 = *(const float4*)(p0);
      float4 a1 = *(const float4*)(p0 + 4);
      float4 b0 = *(const float4*)(p0 + 16);
      float4 b1 = *(const float4*)(p0 + 20);
      BV A; packA(A, a0, a1, b0, b1);
      #pragma unroll
      for (int c = 0; c < 4; ++c) {
        BV Bt;
        Bt.q[0] = *(const u4v*)&muT[16 * c + nlo][kk * 32 + h * 8];
        Bt.q[1] = *(const u4v*)&muT[16 * c + nlo][kk * 32 + h * 8 + 16];
        acc[c] = __builtin_amdgcn_wmma_f32_16x16x32_bf16(
                     false, A.v, false, Bt.v, (short)0, acc[c], false, false);
      }
    }
    __syncthreads();
  }

  // spill pooled C-tiles to this wave's LDS region (C layout: lane=N, vgpr=M)
  #pragma unroll
  for (int c = 0; c < 4; ++c)
    #pragma unroll
    for (int j = 0; j < 8; ++j)
      pool[wave][j + 8 * h][16 * c + nlo] = acc[c][j];

  // mu2 = pool @ mu_2; then two post layers with bias+ReLU (all bf16 WMMA)
  for (int layer = 0; layer < 3; ++layer) {
    v8f y[4] = {};
    #pragma unroll
    for (int kk = 0; kk < 2; ++kk) {
      const float* pr = &pool[wave][nlo][kk * 32 + h * 8];
      float4 a0 = *(const float4*)(pr);
      float4 a1 = *(const float4*)(pr + 4);
      float4 b0 = *(const float4*)(pr + 16);
      float4 b1 = *(const float4*)(pr + 20);
      BV A; packA(A, a0, a1, b0, b1);
      #pragma unroll
      for (int c = 0; c < 4; ++c) {
        BV Bt;
        Bt.q[0] = *(const u4v*)&WT[layer][16 * c + nlo][kk * 32 + h * 8];
        Bt.q[1] = *(const u4v*)&WT[layer][16 * c + nlo][kk * 32 + h * 8 + 16];
        y[c] = __builtin_amdgcn_wmma_f32_16x16x32_bf16(
                   false, A.v, false, Bt.v, (short)0, y[c], false, false);
      }
    }
    if (layer > 0) {
      #pragma unroll
      for (int c = 0; c < 4; ++c) {
        float bv = sBias[layer - 1][16 * c + nlo];
        #pragma unroll
        for (int j = 0; j < 8; ++j) { float vv = y[c][j] + bv; y[c][j] = vv > 0.f ? vv : 0.f; }
      }
    }
    #pragma unroll
    for (int c = 0; c < 4; ++c) {
      #pragma unroll
      for (int j = 0; j < 8; ++j)
        pool[wave][j + 8 * h][16 * c + nlo] = y[c][j];
      acc[c] = y[c];
    }
  }

  // mu = relu(mu_first + mu2); store and produce deterministic partial sums
  const int baseNode = b * N_NODES + r0 + wave * 16;
  float xvw[8];
  #pragma unroll
  for (int j = 0; j < 8; ++j) xvw[j] = xv[baseNode + j + 8 * h];
  #pragma unroll
  for (int c = 0; c < 4; ++c) {
    float part = 0.f;
    const int col = 16 * c + nlo;
    #pragma unroll
    for (int j = 0; j < 8; ++j) {
      size_t idx = (size_t)(baseNode + j + 8 * h) * 64 + col;
      float vv = acc[c][j] + muFirst[idx];
      vv = vv > 0.f ? vv : 0.f;
      muOut[idx] = vv;
      part += xvw[j] * vv;
    }
    wsum[h][wave][col] = part;
  }
  __syncthreads();
  if (tid < 64) {
    float ssum = 0.f;
    #pragma unroll
    for (int w = 0; w < 8; ++w) ssum += wsum[0][w][tid] + wsum[1][w][tid];
    sPart[((size_t)b * 64 + blockIdx.x) * 64 + tid] = ssum;
  }
}

// ---------------------------------------------------------------------------
// Prep: fold linear algebra before the ReLU.
// c1[b] = s[b] @ (q1_w @ W1),  M2 = q2_w @ W2   (W1/W2 = top/bottom of q_reg_w)
// ---------------------------------------------------------------------------
__global__ __launch_bounds__(256) void s2v_prep(
    const float* __restrict__ q1w, const float* __restrict__ q2w,
    const float* __restrict__ qregw, const float* __restrict__ sPart,
    float* __restrict__ M2, float* __restrict__ c1)
{
  __shared__ float M1[64][32];
  __shared__ float sS[2][64];
  const int tid = threadIdx.x;
  for (int idx = tid; idx < 2048; idx += 256) {
    int e = idx >> 5, hc = idx & 31;
    float a1 = 0.f, a2 = 0.f;
    for (int d = 0; d < 64; ++d) {
      a1 += q1w[e * 64 + d] * qregw[d * 32 + hc];
      a2 += q2w[e * 64 + d] * qregw[(64 + d) * 32 + hc];
    }
    M1[e][hc] = a1;
    M2[idx]   = a2;
  }
  if (tid < 128) {
    int bb = tid >> 6, e = tid & 63;
    float s = 0.f;
    for (int g = 0; g < 64; ++g) s += sPart[((size_t)bb * 64 + g) * 64 + e];
    sS[bb][e] = s;
  }
  __syncthreads();
  if (tid < 64) {
    int bb = tid >> 5, hc = tid & 31;
    float cacc = 0.f;
    for (int e = 0; e < 64; ++e) cacc += sS[bb][e] * M1[e][hc];
    c1[tid] = cacc;
  }
}

// ---------------------------------------------------------------------------
// Stage C: out[n] = relu(c1[b] + mu[n] @ M2) @ q_w
// ---------------------------------------------------------------------------
__global__ __launch_bounds__(256) void s2v_out(
    const float* __restrict__ mu, const float* __restrict__ M2,
    const float* __restrict__ c1, const float* __restrict__ qw,
    float* __restrict__ out)
{
  __shared__ float sM2[64][32];
  __shared__ float sc1[2][32];
  __shared__ float sqw[32];
  const int tid = threadIdx.x;
  for (int i = tid; i < 2048; i += 256) ((float*)sM2)[i] = M2[i];
  if (tid < 64) ((float*)sc1)[tid] = c1[tid];
  if (tid < 32) sqw[tid] = qw[tid];
  __syncthreads();

  const int n  = blockIdx.x * 256 + tid;
  const int bb = n >> 13;
  float r[32];
  #pragma unroll
  for (int hc = 0; hc < 32; ++hc) r[hc] = sc1[bb][hc];
  const float* mup = mu + (size_t)n * 64;
  for (int e = 0; e < 64; e += 4) {
    float4 mv = *(const float4*)(mup + e);
    #pragma unroll
    for (int hc = 0; hc < 32; ++hc)
      r[hc] += mv.x * sM2[e][hc] + mv.y * sM2[e + 1][hc] +
               mv.z * sM2[e + 2][hc] + mv.w * sM2[e + 3][hc];
  }
  float o = 0.f;
  #pragma unroll
  for (int hc = 0; hc < 32; ++hc) { float v = r[hc] > 0.f ? r[hc] : 0.f; o += v * sqw[hc]; }
  out[n] = o;
}

// ---------------------------------------------------------------------------
extern "C" void kernel_launch(void* const* d_in, const int* in_sizes, int n_in,
                              void* d_out, int out_size, void* d_ws, size_t ws_size,
                              hipStream_t stream) {
  (void)in_sizes; (void)n_in; (void)out_size; (void)ws_size;
  const float* xv    = (const float*)d_in[0];
  const float* adj   = (const float*)d_in[1];
  const float* mu1   = (const float*)d_in[2];
  const float* mu2w  = (const float*)d_in[3];
  const float* preW  = (const float*)d_in[4];
  const float* preb  = (const float*)d_in[5];
  const float* postW = (const float*)d_in[6];
  const float* postb = (const float*)d_in[7];
  const float* q1w   = (const float*)d_in[8];
  const float* q2w   = (const float*)d_in[9];
  const float* qregw = (const float*)d_in[10];
  const float* qw    = (const float*)d_in[11];

  char* ws = (char*)d_ws;
  float*          muFirst = (float*)(ws + 0);                    // 4,194,304 B
  unsigned short* muPreT  = (unsigned short*)(ws + 4194304);     // 2,097,152 B
  float*          muOut   = (float*)(ws + 6291456);              // 4,194,304 B
  float*          sPart   = (float*)(ws + 10485760);             //   131,072 B
  float*          M2p     = (float*)(ws + 10616832);             //     8,192 B
  float*          c1p     = (float*)(ws + 10625024);             //       256 B

  s2v_embed    <<<dim3(64),    dim3(256), 0, stream>>>(xv, mu1, preW, preb, muFirst, muPreT);
  s2v_aggregate<<<dim3(64, 2), dim3(256), 0, stream>>>(adj, mu2w, postW, postb, xv,
                                                       muFirst, muPreT, muOut, sPart);
  s2v_prep     <<<dim3(1),     dim3(256), 0, stream>>>(q1w, q2w, qregw, sPart, M2p, c1p);
  s2v_out      <<<dim3(64),    dim3(256), 0, stream>>>(muOut, M2p, c1p, qw, (float*)d_out);
}
